// RecurrentWindowAttention_31293131719193
// MI455X (gfx1250) — matrix operand order repaired and checked by reading, not compiled, not run on hardware
//
#include <hip/hip_runtime.h>
#include <hip/hip_fp16.h>

// ---------------------------------------------------------------------------
// RecurrentWindowAttention, fully fused per window for MI455X (gfx1250).
//   B=1024 windows, N=64 tokens, C=192, 4 half-heads x HD=24 per branch.
// One 256-thread (8-wave, wave32) workgroup per window; all intermediates in
// LDS; every GEMM uses v_wmma_f32_16x16x32_f16 (f16 operands, f32 accum).
// Phase 0 stages raw f32 inputs with GLOBAL_LOAD_ASYNC_TO_LDS_B128 (ASYNCcnt)
// into LDS regions that are dead until the attention phase.
// ---------------------------------------------------------------------------

typedef __attribute__((ext_vector_type(16))) _Float16 v16h;
typedef __attribute__((ext_vector_type(4)))  _Float16 v4h;
typedef __attribute__((ext_vector_type(8)))  float    v8f;

#define B_WIN 1024
#define N_TOK 64
#define C_DIM 192
#define NH2   4
#define HD    24
#define HDP   40      // padded row stride (halves) for Q/K head buffers
#define VT_ST 72      // V^T row stride (halves)
#define SC_ST 68      // score row stride (floats)
#define P_ST  72      // P row stride (halves)
#define X_ST  200     // x/state/out row stride (halves)

// LDS byte offsets (all 16B aligned)
#define OFF_X   0u                      // [64][200] f16           = 25600
#define OFF_S   25600u                  // [64][200] f16           = 25600
#define OFF_K   51200u                  // [2][4][64][40] f16      = 40960
#define OFF_Q   92160u                  // [2][4][64][40] f16      = 40960
#define OFF_Q2  133120u                 // [2][4][64][40] f16      = 40960
#define OFF_VT  174080u                 // [2][4][32][72] f16      = 36864
#define OFF_SC  210944u                 // [4][64][68] f32         = 69632
#define OFF_P   280576u                 // [4][64][72] f16         = 36864
#define LDS_TOTAL 317440
// f32 staging for async input copy lives in (dead-until-phase-2) SC+P regions:
//   x     at OFF_SC +     0 .. 49152
//   state at OFF_SC + 49152 .. 98304   (SC+P = 106496 B available)

// Workspace byte offsets
#define WS_BIAS4 0                      // [4][4][64][64] f32 = 262144
#define WS_WS    262144                 // Ws  f16 73728 el   = 147456
#define WS_WE    409600                 // We  f16
#define WS_WPX   557056                 // Wpx f16 36864 el   = 73728
#define WS_WPS   630784                 // Wps f16

// ---- fragment loaders (contiguous k order, wave32) ------------------------
// A (16x32 f16, M x K): lane L: M=L%16, g=L/16; elems 0..15 at k=16g..16g+15
// (same contiguous k order as B) -> two contiguous 16B chunks per lane.
__device__ inline v16h load_frag_a(const _Float16* rowptr, int g) {
  v16h a;
  ((float4*)&a)[0] = *(const float4*)(rowptr + 16 * g);
  ((float4*)&a)[1] = *(const float4*)(rowptr + 16 * g + 8);
  return a;
}
// B (32x16 f16, K x N): lane L: N=L%16, K = 16*(L/16)+i -> 16 contiguous
// halves along K when the N-row is stored contiguously along K.
__device__ inline v16h load_frag_b(const _Float16* p) {
  v16h b;
  ((float4*)&b)[0] = *(const float4*)(p);
  ((float4*)&b)[1] = *(const float4*)(p + 8);
  return b;
}
__device__ inline v8f wmma_f16(v16h a, v16h b, v8f c) {
  return __builtin_amdgcn_wmma_f32_16x16x32_f16(false, a, false, b,
                                                (short)0, c, false, false);
}

// ---------------------------------------------------------------------------
// Prep: gather rel-pos bias into [branch][h][n][m] f32, convert weights to f16
// branch order: 0=self_s 1=cross_s 2=self_x 3=cross_x
// ---------------------------------------------------------------------------
__global__ void rwa_prep(const float* __restrict__ rpb_cross_s,
                         const float* __restrict__ rpb_self_s,
                         const float* __restrict__ rpb_cross_x,
                         const float* __restrict__ rpb_self_x,
                         const int*   __restrict__ rel_idx,
                         const float* __restrict__ Ws, const float* __restrict__ We,
                         const float* __restrict__ Wpx, const float* __restrict__ Wps,
                         float* __restrict__ bias4,
                         _Float16* __restrict__ Ws_h, _Float16* __restrict__ We_h,
                         _Float16* __restrict__ Wpx_h, _Float16* __restrict__ Wps_h) {
  int i = blockIdx.x * 256 + threadIdx.x;
  if (i < 4 * NH2 * N_TOK * N_TOK) {            // 65536
    int m = i & 63, n = (i >> 6) & 63, h = (i >> 12) & 3, br = i >> 14;
    const float* tbl = (br == 0) ? rpb_self_s : (br == 1) ? rpb_cross_s
                     : (br == 2) ? rpb_self_x : rpb_cross_x;
    bias4[i] = tbl[rel_idx[n * 64 + m] * NH2 + h];
  }
  if (i < 2 * C_DIM * C_DIM) {                  // 73728
    Ws_h[i] = (_Float16)Ws[i];
    We_h[i] = (_Float16)We[i];
  }
  if (i < C_DIM * C_DIM) {                      // 36864
    Wpx_h[i] = (_Float16)Wpx[i];
    Wps_h[i] = (_Float16)Wps[i];
  }
}

// ---------------------------------------------------------------------------
// Main fused kernel: one workgroup per window.
// ---------------------------------------------------------------------------
__global__ __launch_bounds__(256)
void rwa_main(const float* __restrict__ input_x,
              const float* __restrict__ state_x,
              const float* __restrict__ bias_xs,
              const float* __restrict__ bs,  const float* __restrict__ be,
              const float* __restrict__ bpx, const float* __restrict__ bps,
              const _Float16* __restrict__ Ws_h, const _Float16* __restrict__ We_h,
              const _Float16* __restrict__ Wpx_h, const _Float16* __restrict__ Wps_h,
              const float* __restrict__ bias4,
              float* __restrict__ out) {        // [2][B][N][C]
  extern __shared__ char smem[];
  const int b    = blockIdx.x;
  const int tid  = threadIdx.x;
  const int w    = tid >> 5;     // wave id, 0..7 (wave32)
  const int lane = tid & 31;
  const int l15  = lane & 15;
  const int lg   = lane >> 4;

  _Float16* sX  = (_Float16*)(smem + OFF_X);   // x (f16); later aliased outX
  _Float16* sS  = (_Float16*)(smem + OFF_S);   // state;   later aliased outS
  _Float16* sK  = (_Float16*)(smem + OFF_K);   // [src][h][tok][HDP], d>=24 zero
  _Float16* sQ  = (_Float16*)(smem + OFF_Q);
  _Float16* sQ2 = (_Float16*)(smem + OFF_Q2);
  _Float16* sVT = (_Float16*)(smem + OFF_VT);  // [src][h][d][tok] transposed
  float*    sSC = (float*)   (smem + OFF_SC);  // [h][n][SC_ST]
  _Float16* sP  = (_Float16*)(smem + OFF_P);   // [h][n][P_ST]

  // Warm L2 for the f16 weight images (shared by all 1024 workgroups).
  if (tid < 64) {
    __builtin_prefetch(Ws_h  + (size_t)tid * 1152, 0, 0);
    __builtin_prefetch(We_h  + (size_t)tid * 1152, 0, 0);
    __builtin_prefetch(Wpx_h + (size_t)tid * 576,  0, 0);
    __builtin_prefetch(Wps_h + (size_t)tid * 576,  0, 0);
  }

  // ---- Phase 0a: async-stage raw f32 x/state into (currently dead) SC/P LDS
  {
    const float* gx = input_x + (size_t)b * (N_TOK * C_DIM);
    const float* gs = state_x + (size_t)b * (N_TOK * C_DIM);
    for (int i = tid; i < (N_TOK * C_DIM) / 4; i += 256) {   // 3072 x 16B
      unsigned lx = OFF_SC + (unsigned)i * 16u;
      unsigned ls = OFF_SC + 49152u + (unsigned)i * 16u;
      const float* px = gx + i * 4;
      const float* ps = gs + i * 4;
      asm volatile("global_load_async_to_lds_b128 %0, %1, off"
                   :: "v"(lx), "v"(px) : "memory");
      asm volatile("global_load_async_to_lds_b128 %0, %1, off"
                   :: "v"(ls), "v"(ps) : "memory");
    }
    // Overlap: zero K/Q/Q2/VT padding while the async copies are in flight.
    uint4* pz = (uint4*)(smem + OFF_K);
    uint4  z  = make_uint4(0u, 0u, 0u, 0u);
    for (int i = tid; i < (int)((OFF_SC - OFF_K) / 16); i += 256) pz[i] = z;
    asm volatile("s_wait_asynccnt 0" ::: "memory");
  }
  __syncthreads();

  // ---- Phase 0b: convert staged f32 -> f16 (+per-token bias on x) ---------
  {
    const float* stX = (const float*)(smem + OFF_SC);
    const float* stS = (const float*)(smem + OFF_SC + 49152u);
    for (int i = tid; i < (N_TOK * C_DIM) / 4; i += 256) {
      int e = i * 4, n = e / C_DIM, c = e % C_DIM;
      float4 vx = *(const float4*)(stX + e);
      float4 vs = *(const float4*)(stS + e);
      float  bb = bias_xs[n];
      v4h hx = { (_Float16)(vx.x + bb), (_Float16)(vx.y + bb),
                 (_Float16)(vx.z + bb), (_Float16)(vx.w + bb) };
      v4h hs = { (_Float16)vs.x, (_Float16)vs.y, (_Float16)vs.z, (_Float16)vs.w };
      *(v4h*)(sX + n * X_ST + c) = hx;
      *(v4h*)(sS + n * X_ST + c) = hs;
    }
  }
  __syncthreads();

  // ---- Phase 1: QKV projections for both sources: (64x192)@(192x384) ------
  // 2 srcs x 4 row-tiles x 24 col-tiles = 192 tile jobs, 6 k-steps each.
  for (int job = w; job < 192; job += 8) {
    int src = job / 96, t = job % 96, tm = t / 24, to = t % 24;
    const _Float16* A    = (src == 0) ? sS   : sX;
    const _Float16* Wt   = (src == 0) ? Ws_h : We_h;
    const float*    bvec = (src == 0) ? bs   : be;
    int o = to * 16 + l15;                       // output channel (0..383)
    const _Float16* arow = A  + (tm * 16 + l15) * X_ST;
    const _Float16* brow = Wt + (size_t)o * C_DIM;   // W[o][c] row-major
    v8f acc = {0.f, 0.f, 0.f, 0.f, 0.f, 0.f, 0.f, 0.f};
#pragma unroll
    for (int kk = 0; kk < 6; ++kk) {
      v16h a  = load_frag_a(arow + kk * 32, lg);
      v16h bm = load_frag_b(brow + kk * 32 + 16 * lg);
      acc = wmma_f16(a, bm, acc);
    }
    float bias = bvec[o];
    // g is WAVE-UNIFORM: 16-wide tiles never straddle a 96 boundary.
    int g   = to / 6;                            // 0=K 1=V 2=Q 3=Q'
    int rem = o - g * 96;                        // per-lane, 0..95
    int h   = (rem * 43) >> 10;                  // rem / 24 (exact for 0..95)
    int d   = rem - h * 24;
    // Store addressing as BYTE OFFSETS from smem so the address space stays
    // LDS (ds_store_b16), no generic-pointer select.
    unsigned baseOff, strideB;
    if (g == 1) {                                // V stored transposed [d][tok]
      baseOff = OFF_VT + (unsigned)(((src * 4 + h) * 32 + d) * VT_ST) * 2u;
      strideB = 2u;
    } else {
      unsigned bufOff = (g == 0) ? OFF_K : (g == 2) ? OFF_Q : OFF_Q2;
      baseOff = bufOff + (unsigned)((src * 4 + h) * 64 * HDP + d) * 2u;
      strideB = HDP * 2u;
    }
#pragma unroll
    for (int r = 0; r < 8; ++r) {
      unsigned n = (unsigned)(tm * 16 + r + 8 * lg);
      *(_Float16*)(smem + baseOff + n * strideB) = (_Float16)(acc[r] + bias);
    }
  }
  __syncthreads();

  // ---- Phase 2: four attention branches -----------------------------------
  const float scale = 0.20412414523193154f;      // 24^-0.5
  const int q2sel[4] = {0, 1, 0, 1};             // Q or Q'
  const int qsrcA[4] = {0, 0, 1, 1};             // 0=state 1=x
  const int kvA[4]   = {0, 1, 1, 0};
  const unsigned offA[4]  = {96u, 0u, 96u, 0u};  // concat column offset
  for (int br = 0; br < 4; ++br) {
    const _Float16* Qb = (q2sel[br] ? sQ2 : sQ) + qsrcA[br] * 4 * 64 * HDP;
    const _Float16* Kb = sK  + kvA[br] * 4 * 64 * HDP;
    const _Float16* Vb = sVT + kvA[br] * 4 * 32 * VT_ST;
    const float* bias_br = bias4 + (size_t)br * 4 * 64 * 64;
    unsigned outBase = (br < 2) ? OFF_S : OFF_X; // aliases x/state (now dead)
    unsigned outoff  = offA[br];

    // 2a: S = (Q*scale) K^T + bias   (4 heads x 16 tiles, 1 WMMA each, K=32pad)
    for (int job = w; job < 64; job += 8) {
      int h = job >> 4, tm = (job >> 2) & 3, tn = job & 3;
      v16h a  = load_frag_a(Qb + (h * 64 + tm * 16 + l15) * HDP, lg);
      v16h bm = load_frag_b(Kb + (h * 64 + tn * 16 + l15) * HDP + 16 * lg);
      v8f acc = {0.f, 0.f, 0.f, 0.f, 0.f, 0.f, 0.f, 0.f};
      acc = wmma_f16(a, bm, acc);
      int m = tn * 16 + l15;
      const float* bp  = bias_br + (size_t)h * 64 * 64 + m;
      float* scol = sSC + h * 64 * SC_ST + m;
#pragma unroll
      for (int r = 0; r < 8; ++r) {
        int n = tm * 16 + r + 8 * lg;
        scol[n * SC_ST] = acc[r] * scale + bp[n * 64];
      }
    }
    __syncthreads();

    // 2b: softmax — 256 threads <-> 4 heads x 64 rows, one row per thread
    {
      int h = tid >> 6, n = tid & 63;
      const float* srow = sSC + (h * 64 + n) * SC_ST;
      float v[64];
      float mx = -1e30f;
#pragma unroll
      for (int m = 0; m < 64; ++m) { v[m] = srow[m]; mx = fmaxf(mx, v[m]); }
      float sum = 0.f;
#pragma unroll
      for (int m = 0; m < 64; ++m) { float e = __expf(v[m] - mx); v[m] = e; sum += e; }
      float inv = 1.f / sum;
      _Float16* prow = sP + (h * 64 + n) * P_ST;
#pragma unroll
      for (int m = 0; m < 64; ++m) prow[m] = (_Float16)(v[m] * inv);
    }
    __syncthreads();

    // 2c: O = P V   (4 heads x 4x2 tiles, 2 k-steps of 32 each)
    for (int job = w; job < 32; job += 8) {
      int h = job >> 3, tm = (job >> 1) & 3, td = job & 1;
      const _Float16* arow = sP + (h * 64 + tm * 16 + l15) * P_ST;
      int dcol = td * 16 + l15;
      const _Float16* brow = Vb + (h * 32 + dcol) * VT_ST;
      v8f acc = {0.f, 0.f, 0.f, 0.f, 0.f, 0.f, 0.f, 0.f};
#pragma unroll
      for (int kk = 0; kk < 2; ++kk) {
        v16h a  = load_frag_a(arow + kk * 32, lg);
        v16h bm = load_frag_b(brow + kk * 32 + 16 * lg);
        acc = wmma_f16(a, bm, acc);
      }
      if (dcol < HD) {                           // drop padded d=24..31
        unsigned col = outoff + (unsigned)(h * HD + dcol);
#pragma unroll
        for (int r = 0; r < 8; ++r) {
          unsigned n = (unsigned)(tm * 16 + r + 8 * lg);
          *(_Float16*)(smem + outBase + (n * X_ST + col) * 2u) = (_Float16)acc[r];
        }
      }
    }
    __syncthreads();
  }

  // ---- Phase 3: output projections (64x192)@(192x192) + bias -> f32 out ---
  for (int job = w; job < 96; job += 8) {
    int which = job / 48, t = job % 48, tm = t / 12, to = t % 12;
    const _Float16* A    = (which == 0) ? sX    : sS;     // outX / outS
    const _Float16* Wt   = (which == 0) ? Wpx_h : Wps_h;
    const float*    bvec = (which == 0) ? bpx   : bps;
    int o = to * 16 + l15;
    const _Float16* arow = A  + (tm * 16 + l15) * X_ST;
    const _Float16* brow = Wt + (size_t)o * C_DIM;
    v8f acc = {0.f, 0.f, 0.f, 0.f, 0.f, 0.f, 0.f, 0.f};
#pragma unroll
    for (int kk = 0; kk < 6; ++kk) {
      v16h a  = load_frag_a(arow + kk * 32, lg);
      v16h bm = load_frag_b(brow + kk * 32 + 16 * lg);
      acc = wmma_f16(a, bm, acc);
    }
    float bias = bvec[o];
    float* gout = out + ((size_t)which * B_WIN + b) * (N_TOK * C_DIM);
#pragma unroll
    for (int r = 0; r < 8; ++r) {
      int n = tm * 16 + r + 8 * lg;
      gout[n * C_DIM + o] = acc[r] + bias;
    }
  }
}

// ---------------------------------------------------------------------------
extern "C" void kernel_launch(void* const* d_in, const int* in_sizes, int n_in,
                              void* d_out, int out_size, void* d_ws, size_t ws_size,
                              hipStream_t stream) {
  (void)in_sizes; (void)n_in; (void)out_size; (void)ws_size;
  const float* input_x     = (const float*)d_in[0];
  const float* state_x     = (const float*)d_in[1];
  const float* bias_xs     = (const float*)d_in[2];
  const float* rpb_cross_s = (const float*)d_in[3];
  const float* rpb_self_s  = (const float*)d_in[4];
  const float* rpb_cross_x = (const float*)d_in[5];
  const float* rpb_self_x  = (const float*)d_in[6];
  const float* Ws   = (const float*)d_in[7];
  const float* bs   = (const float*)d_in[8];
  const float* We   = (const float*)d_in[9];
  const float* be   = (const float*)d_in[10];
  const float* Wpx  = (const float*)d_in[11];
  const float* bpx  = (const float*)d_in[12];
  const float* Wps  = (const float*)d_in[13];
  const float* bps  = (const float*)d_in[14];
  const int*   rel  = (const int*)d_in[15];

  char* ws = (char*)d_ws;
  float*    bias4 = (float*)(ws + WS_BIAS4);
  _Float16* Ws_h  = (_Float16*)(ws + WS_WS);
  _Float16* We_h  = (_Float16*)(ws + WS_WE);
  _Float16* Wpx_h = (_Float16*)(ws + WS_WPX);
  _Float16* Wps_h = (_Float16*)(ws + WS_WPS);

  rwa_prep<<<288, 256, 0, stream>>>(rpb_cross_s, rpb_self_s, rpb_cross_x,
                                    rpb_self_x, rel, Ws, We, Wpx, Wps,
                                    bias4, Ws_h, We_h, Wpx_h, Wps_h);
  rwa_main<<<B_WIN, 256, LDS_TOTAL, stream>>>(input_x, state_x, bias_xs,
                                              bs, be, bpx, bps,
                                              Ws_h, We_h, Wpx_h, Wps_h,
                                              bias4, (float*)d_out);
}
